// QuantumNeuralNetwork_10831907520868
// MI455X (gfx1250) — matrix operand order repaired and checked by reading, not compiled, not run on hardware
//
#include <hip/hip_runtime.h>

// MI455X (gfx1250) 16-qubit statevector QNN, multi-pass L2-resident design.
// State: B x 65536 float2 in d_ws (32 MB, fits 192 MB L2). 13 streaming passes.
// Qubit q == amplitude bit (15-q) (reference reshape semantics).
//
// Low kernel:  one 2048-amp tile (amp bits 0-10) per block in LDS
//              (async global<->LDS copies, ASYNCcnt-tracked).
//              - composed CNOT(10,9)..(1,0) permutation as one LDS gather
//              - RX on bits 0-3 and 4-7 fused into 16x16 complex matrices,
//                applied with split-f16 (hi+lo) v_wmma_f32_16x16x32_f16:
//                D ~= Ah*Bh + Ah*Bl + Al*Bh  (f32 accumulate, ~2^-22 error)
//              - RX bits 8,9 as LDS butterflies
// High kernel: 64 amps/thread (amp bits 10-15) in VGPRs.
// Measurement: sign (-1)^(bit15^bit0) folds the final wrap CNOT.

#define USE_ASYNC_LDS 1

typedef __attribute__((ext_vector_type(16))) _Float16 v16h;
typedef __attribute__((ext_vector_type(8)))  float    v8f;

// ---------- RX pair rotation: a'=c*a - i*s*b ; b'=-i*s*a + c*b ----------
__device__ __forceinline__ void rot(float2& a, float2& b, float c, float s) {
  float2 na = make_float2(c * a.x + s * b.y, c * a.y - s * b.x);
  float2 nb = make_float2(c * b.x + s * a.y, c * b.y - s * a.x);
  a = na; b = nb;
}

__device__ __forceinline__ void sincos_fast(float a, float& s, float& c) {
  __sincosf(a, &s, &c);   // v_sin_f32 / v_cos_f32, no Payne-Hanek SALU bloat
}

// ---------- fused 4-qubit RX block: A = [Gr|-Gi], [Gi|Gr], split f16 --------
// G(m,k) = (-i)^popc(m^k) * prod_j (bit_j(m^k) ? sin_j : cos_j), angles th4[j]
__device__ __forceinline__ void build_A(const float* th4, int lane,
                                        v16h& arh, v16h& arl,
                                        v16h& aih, v16h& ail) {
  float cq[4], sq[4];
#pragma unroll
  for (int j = 0; j < 4; ++j) sincos_fast(0.5f * th4[j], sq[j], cq[j]);
  const int m = lane & 15;
  float gr[16], gi[16];
#pragma unroll
  for (int k = 0; k < 16; ++k) {
    const int xk = m ^ k;
    float p = 1.f;
#pragma unroll
    for (int j = 0; j < 4; ++j) p *= ((xk >> j) & 1) ? sq[j] : cq[j];
    const int d = __popc(xk) & 3;
    gr[k] = (d == 0) ? p : ((d == 2) ? -p : 0.f);
    gi[k] = (d == 3) ? p : ((d == 1) ? -p : 0.f);
  }
  // A layout (16x32 f16): lane<16 holds K = e, lane>=16 holds K = e+16.
  // Select between the two compile-time K options via ternary (no dyn index).
  const bool hi = (lane & 16) != 0;
#pragma unroll
  for (int e = 0; e < 16; ++e) {
    const int K0 = e;
    const int K1 = e + 16;
    const float r0 = gr[K0 & 15], i0 = gi[K0 & 15];
    const float r1 = gr[K1 & 15], i1 = gi[K1 & 15];
    const float arv = hi ? ((K1 >= 16) ? -i1 : r1) : ((K0 >= 16) ? -i0 : r0);
    const float aiv = hi ? ((K1 >= 16) ? r1 : i1) : ((K0 >= 16) ? r0 : i0);
    const _Float16 ah = (_Float16)arv;
    const _Float16 bh = (_Float16)aiv;
    arh[e] = ah; arl[e] = (_Float16)(arv - (float)ah);
    aih[e] = bh; ail[e] = (_Float16)(aiv - (float)bh);
  }
}

// ---------- apply fused 16x16 complex gate to the LDS tile via WMMA ----------
// VAR=0: vector index = amp bits 0-3 (16 consecutive amps per group)
// VAR=1: vector index = amp bits 4-7 (stride-16 groups)
// Wave wv owns LDS region [wv*256, wv*256+256) in both variants -> no races.
template<int VAR>
__device__ __forceinline__ void wmma_apply(float2* tile, const float* th4,
                                           int lane, int wv) {
  v16h arh, arl, aih, ail;
  build_A(th4, lane, arh, arl, aih, ail);
  const int n = lane & 15;
  const int g = wv * 16 + n;
  v16h bh, bl;  // B (32x16): lanes 0-15 supply Sr rows, lanes 16-31 Si rows
#pragma unroll
  for (int e = 0; e < 16; ++e) {
    const int idx = (VAR == 0) ? (g * 16 + e) : ((wv << 8) | (e << 4) | n);
    const float2 v = tile[idx];
    const float vs = (lane < 16) ? v.x : v.y;
    const _Float16 h = (_Float16)vs;
    bh[e] = h;
    bl[e] = (_Float16)(vs - (float)h);
  }
  v8f dr = {}, di = {};
  dr = __builtin_amdgcn_wmma_f32_16x16x32_f16(false, arh, false, bh, (short)0, dr, false, false);
  dr = __builtin_amdgcn_wmma_f32_16x16x32_f16(false, arh, false, bl, (short)0, dr, false, false);
  dr = __builtin_amdgcn_wmma_f32_16x16x32_f16(false, arl, false, bh, (short)0, dr, false, false);
  di = __builtin_amdgcn_wmma_f32_16x16x32_f16(false, aih, false, bh, (short)0, di, false, false);
  di = __builtin_amdgcn_wmma_f32_16x16x32_f16(false, aih, false, bl, (short)0, di, false, false);
  di = __builtin_amdgcn_wmma_f32_16x16x32_f16(false, ail, false, bh, (short)0, di, false, false);
  const int mb = (lane & 16) ? 8 : 0;
#pragma unroll
  for (int j = 0; j < 8; ++j) {
    const int mm = mb + j;
    const int idx = (VAR == 0) ? (g * 16 + mm) : ((wv << 8) | (mm << 4) | n);
    tile[idx] = make_float2(dr[j], di[j]);
  }
}

// ---------- LOW kernel: amp bits 0-10 inside a 2048-amp LDS tile ----------
// perm==0 (init): synthesize |0..0>, th[b] = x[b,15-b] + w[0,15-b], no CNOTs.
// perm==1 (layer l): gather-apply CNOT(10,9)..(1,0) of layer l, then
//   RX bit0 angle w[l,15] (l>=1), RX bits1-9 angles w[l+1,14-..] (l+1<L).
__global__ void __launch_bounds__(256)
qnn_low(float2* st, const float* __restrict__ x, const float* __restrict__ w,
        int perm, int l, int L) {
  __shared__ __align__(16) float2 tile[2048];
  const int t = threadIdx.x;
  const int tb = blockIdx.x;                  // tile 0..31 (amp bits 11-15)
  const int b = blockIdx.y;                   // batch
  const size_t base = (size_t)b * 65536 + (size_t)tb * 2048;
  const int lane = t & 31;
  const int wv = t >> 5;

  float th[10];
  if (!perm) {
#pragma unroll
    for (int bit = 0; bit < 10; ++bit)
      th[bit] = x[b * 16 + (15 - bit)] + w[15 - bit];
  } else {
    th[0] = (l >= 1) ? w[l * 16 + 15] : 0.f;
#pragma unroll
    for (int bit = 1; bit < 10; ++bit)
      th[bit] = (l + 1 < L) ? w[(l + 1) * 16 + (15 - bit)] : 0.f;
  }

  // load (or synthesize |0>)
  if (!perm) {
#pragma unroll
    for (int j = 0; j < 8; ++j) {
      const int idx = t + 256 * j;
      tile[idx] = make_float2((tb == 0 && idx == 0) ? 1.f : 0.f, 0.f);
    }
  } else {
#if USE_ASYNC_LDS
#pragma unroll
    for (int j = 0; j < 4; ++j) {
      const int g16 = t + 256 * j;              // 16-byte granule index
      asm volatile("global_load_async_to_lds_b128 %0, %1, off"
                   :: "v"((unsigned)(size_t)(tile + g16 * 2)),
                      "v"(st + base + g16 * 2)
                   : "memory");
    }
    asm volatile("s_wait_asynccnt 0x0" ::: "memory");
#else
#pragma unroll
    for (int j = 0; j < 8; ++j) {
      const int idx = t + 256 * j;
      tile[idx] = st[base + idx];
    }
#endif
  }
  __syncthreads();

  if (perm) {
    // composed permutation of CNOT(10,9),(9,8),...,(1,0):
    // src = apply last-gate-first: c = 1..10 : s ^= bit_c(s) << (c-1)
    float2 tmp[8];
#pragma unroll
    for (int j = 0; j < 8; ++j) {
      const int d = t + 256 * j;
      int s = d;
#pragma unroll
      for (int c = 1; c <= 10; ++c) s ^= ((s >> c) & 1) << (c - 1);
      tmp[j] = tile[s];
    }
    __syncthreads();
#pragma unroll
    for (int j = 0; j < 8; ++j) tile[t + 256 * j] = tmp[j];
    __syncthreads();
  }

  // fused RX on amp bits 0-3 and 4-7 (split-f16 WMMA)
  wmma_apply<0>(tile, &th[0], lane, wv);
  __syncthreads();
  wmma_apply<1>(tile, &th[4], lane, wv);
  __syncthreads();

  // RX butterflies on amp bits 8 and 9
  {
    float s8, c8; sincos_fast(0.5f * th[8], s8, c8);
#pragma unroll
    for (int j = 0; j < 4; ++j) {
      const int p = t + 256 * j;
      const int i0 = ((p >> 8) << 9) | (p & 255);
      rot(tile[i0], tile[i0 | 256], c8, s8);
    }
  }
  __syncthreads();
  {
    float s9, c9; sincos_fast(0.5f * th[9], s9, c9);
#pragma unroll
    for (int j = 0; j < 4; ++j) {
      const int p = t + 256 * j;
      const int i0 = ((p >> 9) << 10) | (p & 511);
      rot(tile[i0], tile[i0 | 512], c9, s9);
    }
  }
  __syncthreads();

#if USE_ASYNC_LDS
#pragma unroll
  for (int j = 0; j < 4; ++j) {
    const int g16 = t + 256 * j;
    asm volatile("global_store_async_from_lds_b128 %0, %1, off"
                 :: "v"(st + base + g16 * 2),
                    "v"((unsigned)(size_t)(tile + g16 * 2))
                 : "memory");
  }
  // s_endpgm performs an implicit wait-idle; stores complete before kernel end
#else
#pragma unroll
  for (int j = 0; j < 8; ++j) st[base + t + 256 * j] = tile[t + 256 * j];
#endif
}

// ---------- HIGH kernel helpers: in-register gates on m = amp bits 10-15 ----
template<int M>
__device__ __forceinline__ void hrot(float2 (&r)[64], float th) {
  float s, c; sincos_fast(0.5f * th, s, c);
#pragma unroll
  for (int m = 0; m < 64; ++m)
    if (!(m & M)) rot(r[m], r[m | M], c, s);
}

template<int CM, int TM>
__device__ __forceinline__ void hcnot(float2 (&r)[64]) {
#pragma unroll
  for (int m = 0; m < 64; ++m)
    if ((m & CM) && !(m & TM)) {
      float2 tmp = r[m]; r[m] = r[m | TM]; r[m | TM] = tmp;
    }
}

// mode==0: RX(x+w0, bits 10-15) then CNOT(15,14)..(11,10)
// mode==1: wrap CNOT(0,15) of layer lp-1, RX(w_lp, bits 10-15), CNOT hi chain
__global__ void __launch_bounds__(256)
qnn_high(float2* st, const float* __restrict__ x, const float* __restrict__ w,
         int mode, int lp) {
  const int t = threadIdx.x;
  const int f = blockIdx.x * 256 + t;         // fixed amp bits 0-9
  const int b = blockIdx.y;
  const size_t base = (size_t)b * 65536 + (size_t)f;

  float2 r[64];
#pragma unroll
  for (int m = 0; m < 64; ++m) r[m] = st[base + (size_t)m * 1024];

  float th[6];
  if (mode == 0) {
#pragma unroll
    for (int k = 0; k < 6; ++k) th[k] = x[b * 16 + (5 - k)] + w[5 - k];
  } else {
    // CNOT(control = amp bit0 = f&1, target = amp bit15 = m bit5)
    if (f & 1) {
#pragma unroll
      for (int m = 0; m < 32; ++m) {
        float2 tmp = r[m]; r[m] = r[m + 32]; r[m + 32] = tmp;
      }
    }
#pragma unroll
    for (int k = 0; k < 6; ++k) th[k] = w[lp * 16 + (5 - k)];
  }

  hrot<1>(r, th[0]);  hrot<2>(r, th[1]);  hrot<4>(r, th[2]);
  hrot<8>(r, th[3]);  hrot<16>(r, th[4]); hrot<32>(r, th[5]);

  // CNOT (15,14),(14,13),(13,12),(12,11),(11,10) in order (m-bit pairs)
  hcnot<32, 16>(r); hcnot<16, 8>(r); hcnot<8, 4>(r);
  hcnot<4, 2>(r);   hcnot<2, 1>(r);

#pragma unroll
  for (int m = 0; m < 64; ++m) st[base + (size_t)m * 1024] = r[m];
}

// ---------- measurement: <Z_q0>, sign folds the final wrap CNOT ----------
__global__ void __launch_bounds__(256)
qnn_meas(const float2* __restrict__ st, float* __restrict__ part) {
  __shared__ float red[256];
  const int t = threadIdx.x, tb = blockIdx.x, b = blockIdx.y;
  const size_t base = (size_t)b * 65536 + (size_t)tb * 2048;
  const int bit15 = (tb >> 4) & 1;
  float acc = 0.f;
#pragma unroll
  for (int j = 0; j < 8; ++j) {
    const int idx = t + 256 * j;
    const float2 a = st[base + idx];
    const float sgn = ((bit15 ^ (idx & 1)) ? -1.f : 1.f);
    acc += sgn * (a.x * a.x + a.y * a.y);
  }
  red[t] = acc;
  __syncthreads();
#pragma unroll
  for (int s = 128; s > 0; s >>= 1) {
    if (t < s) red[t] += red[t + s];
    __syncthreads();
  }
  if (t == 0) part[b * 32 + tb] = red[0];
}

__global__ void qnn_reduce(const float* __restrict__ part,
                           float* __restrict__ out, int B) {
  const int b = blockIdx.x * blockDim.x + threadIdx.x;
  if (b < B) {
    float s = 0.f;
#pragma unroll
    for (int i = 0; i < 32; ++i) s += part[b * 32 + i];
    out[b] = s;
  }
}

extern "C" void kernel_launch(void* const* d_in, const int* in_sizes, int n_in,
                              void* d_out, int out_size, void* d_ws, size_t ws_size,
                              hipStream_t stream) {
  const float* x = (const float*)d_in[0];   // (B,16) f32
  const float* w = (const float*)d_in[1];   // (L,16) f32
  float* out = (float*)d_out;               // (B,) f32
  const int B = in_sizes[0] / 16;
  const int L = in_sizes[1] / 16;
  (void)n_in; (void)out_size; (void)ws_size;

  float2* st = (float2*)d_ws;                                   // B*65536 cplx
  float* part = (float*)((char*)d_ws + (size_t)B * 65536 * sizeof(float2));

  const dim3 blk(256);
  // init + all-RX of encoding and layer 0 (angles add on the same qubit)
  qnn_low <<<dim3(32, B), blk, 0, stream>>>(st, x, w, 0, 0, L);
  qnn_high<<<dim3(4, B),  blk, 0, stream>>>(st, x, w, 0, 0);
  for (int l = 0; l < L; ++l) {
    qnn_low <<<dim3(32, B), blk, 0, stream>>>(st, x, w, 1, l, L);
    if (l + 1 < L)
      qnn_high<<<dim3(4, B), blk, 0, stream>>>(st, x, w, 1, l + 1);
  }
  qnn_meas  <<<dim3(32, B), blk, 0, stream>>>(st, part);
  qnn_reduce<<<dim3((B + 63) / 64), dim3(64), 0, stream>>>(part, out, B);
}
